// AttentionCroiseeVariables_83915071029419
// MI455X (gfx1250) — matrix operand order repaired and checked by reading, not compiled, not run on hardware
//
#include <hip/hip_runtime.h>
#include <hip/hip_bf16.h>

typedef __bf16 bf16_t;
typedef __attribute__((ext_vector_type(16))) __bf16 v16bf;
typedef __attribute__((ext_vector_type(8)))  __bf16 v8bf;
typedef __attribute__((ext_vector_type(8)))  float  v8f;
typedef __attribute__((vector_size(16)))     int    vi4;

#define NV 4
#define BB 4
#define TT 512
#define DD 512
#define HH 8
#define DK 64

#if __has_builtin(__builtin_amdgcn_global_load_async_to_lds_b128)
#define USE_ASYNC_LDS 1
#else
#define USE_ASYNC_LDS 0
#endif

// ---------- WMMA helpers ----------
__device__ __forceinline__ v8f wmma_bf16(v16bf a, v16bf b, v8f c) {
  return __builtin_amdgcn_wmma_f32_16x16x32_bf16(false, a, false, b, (short)0, c,
                                                 false, false);
}

// A-fragment (16x32) from row-major: p = &M[row = lane%16][kbase + 16*(lane/16)], contiguous
__device__ __forceinline__ v16bf afrag(const bf16_t* p) {
  v8bf lo = *(const v8bf*)p;
  v8bf hi = *(const v8bf*)(p + 8);
  v16bf a;
#pragma unroll
  for (int i = 0; i < 8; ++i) { a[i] = lo[i]; a[8 + i] = hi[i]; }
  return a;
}

// B-fragment (32x16): p = &M[row = n0 + lane%16][kbase + 16*(lane/16)], contiguous
__device__ __forceinline__ v16bf bfrag(const bf16_t* p) {
  return *(const v16bf*)p;
}

// 16-byte global->LDS copy: async op when available, else staged copy.
__device__ __forceinline__ void copy16(const bf16_t* g, bf16_t* l) {
#if USE_ASYNC_LDS
  __builtin_amdgcn_global_load_async_to_lds_b128((vi4*)g, (vi4*)l, 0, 0);
#else
  *(uint4*)l = *(const uint4*)g;
#endif
}

__device__ __forceinline__ void wait_async() {
#if USE_ASYNC_LDS
#if __has_builtin(__builtin_amdgcn_s_wait_asynccnt)
  __builtin_amdgcn_s_wait_asynccnt(0);
#else
  asm volatile("s_wait_asynccnt 0x0" ::: "memory");
#endif
#endif
}

// ---------- conversion kernels ----------
__global__ void cvt_bf16(const float* __restrict__ in, bf16_t* __restrict__ out, int n) {
  int i = blockIdx.x * 256 + threadIdx.x;
  if (i < n) out[i] = (bf16_t)in[i];
}

// W [512x512] -> WT bf16, WT[n*512+k] = W[k*512+n]
__global__ void cvt_bf16_T(const float* __restrict__ in, bf16_t* __restrict__ out) {
  int i = blockIdx.x * 256 + threadIdx.x;
  int n = i >> 9, k = i & 511;
  out[i] = (bf16_t)in[(size_t)k * 512 + n];
}

// ---------- WMMA GEMM, 32x64 tile per wave: C[M x 512] = A * BT^T + bias ----------
// mode 0: bf16 row-major (Q,K) | mode 1: bf16 V transposed | mode 2: f32 (out proj)
__global__ void gemm_bf16(const bf16_t* __restrict__ A, const bf16_t* __restrict__ BT,
                          const float* __restrict__ bias, void* __restrict__ out,
                          int mode) {
  const int lane = threadIdx.x & 31;
  const int half = lane >> 4;
  const int l16  = lane & 15;
  const int tn = blockIdx.x & 7;        // 8 col groups of 64
  const int tm = blockIdx.x >> 3;       // M/32 row groups
  const int row = tm * 32;
  const int n0  = tn * 64;

  v8f acc[2][4] = {};
  const bf16_t* ap0 = A + (size_t)(row + l16) * 512 + 16 * half;
  const bf16_t* ap1 = ap0 + (size_t)16 * 512;
  const bf16_t* bp  = BT + (size_t)(n0 + l16) * 512 + 16 * half;
#pragma unroll 4
  for (int kk = 0; kk < 512; kk += 32) {
    v16bf a0 = afrag(ap0 + kk);
    v16bf a1 = afrag(ap1 + kk);
#pragma unroll
    for (int j = 0; j < 4; ++j) {
      v16bf bm = bfrag(bp + (size_t)j * 16 * 512 + kk);
      acc[0][j] = wmma_bf16(a0, bm, acc[0][j]);
      acc[1][j] = wmma_bf16(a1, bm, acc[1][j]);
    }
  }

  float bvj[4];
#pragma unroll
  for (int j = 0; j < 4; ++j) bvj[j] = bias[n0 + j * 16 + l16];

  if (mode == 2) {
    float* o = (float*)out;
#pragma unroll
    for (int i = 0; i < 2; ++i)
#pragma unroll
      for (int j = 0; j < 4; ++j)
#pragma unroll
        for (int r = 0; r < 8; ++r) {
          int m = row + i * 16 + r + 8 * half;
          o[(size_t)m * 512 + n0 + j * 16 + l16] = acc[i][j][r] + bvj[j];
        }
  } else if (mode == 0) {
    bf16_t* o = (bf16_t*)out;
#pragma unroll
    for (int i = 0; i < 2; ++i)
#pragma unroll
      for (int j = 0; j < 4; ++j)
#pragma unroll
        for (int r = 0; r < 8; ++r) {
          int m = row + i * 16 + r + 8 * half;
          o[(size_t)m * 512 + n0 + j * 16 + l16] = (bf16_t)(acc[i][j][r] + bvj[j]);
        }
  } else {
    bf16_t* o = (bf16_t*)out;
#pragma unroll
    for (int i = 0; i < 2; ++i)
#pragma unroll
      for (int j = 0; j < 4; ++j) {
        const int ng = n0 + j * 16 + l16, h = ng >> 6, nn = ng & 63;
#pragma unroll
        for (int r = 0; r < 8; ++r) {
          int m  = row + i * 16 + r + 8 * half;   // flat t over N*B*T
          int cb = m >> 9, t = m & 511;
          o[((size_t)(cb * 8 + h) * 64 + nn) * 512 + t] = (bf16_t)(acc[i][j][r] + bvj[j]);
        }
      }
  }
}

// ---------- flash attention: 4 waves/WG share K/V tiles via (async) LDS ----------
// grid: q(4)*b(4)*h(8)*rowgroups(8) = 1024 blocks of 128 threads
#define PSTRIDE 40
__global__ void attn_kernel(const bf16_t* __restrict__ Q, const bf16_t* __restrict__ K,
                            const bf16_t* __restrict__ Vt, bf16_t* __restrict__ ctxh) {
  __shared__ __align__(32) bf16_t ldsK[2][32 * 64];   // [buf][key 32][dk 64]
  __shared__ __align__(32) bf16_t ldsV[2][64 * 32];   // [buf][dv 64][key 32]
  __shared__ __align__(32) bf16_t pshm[4][16 * PSTRIDE];

  const int tid  = threadIdx.x;
  const int wave = tid >> 5;
  const int lane = tid & 31;
  const int half = lane >> 4;
  const int l16  = lane & 15;
  int wg = blockIdx.x;
  const int rg = wg & 7; wg >>= 3;
  const int h  = wg & 7; wg >>= 3;
  const int b  = wg & 3; wg >>= 2;
  const int q  = wg;
  const int t0 = (rg * 4 + wave) * 16;
  const size_t qrow = (size_t)(q * BB + b) * TT + t0;

  const bf16_t* qp = Q + (qrow + l16) * 512 + h * DK + 16 * half;
  const v16bf qf0 = afrag(qp);
  const v16bf qf1 = afrag(qp + 32);

  v8f acc0 = {}, acc1 = {}, acc2 = {}, acc3 = {};
  bf16_t* pbase = &pshm[wave][0];

  for (int c = 0; c < NV; ++c) {
    const size_t krow   = (size_t)(c * BB + b) * TT;
    const size_t vtbase = (size_t)((c * BB + b) * HH + h) * DK;
    v8f O0 = {}, O1 = {}, O2 = {}, O3 = {};
    float mrow[8], lrow[8];
#pragma unroll
    for (int r = 0; r < 8; ++r) { mrow[r] = -1e30f; lrow[r] = 0.f; }

    // cooperative tile fetch: 512 16B chunks over 128 threads (4 each), uniform split
    auto issue = [&](int s0, int buf) {
#pragma unroll
      for (int j = 0; j < 4; ++j) {
        int cidx = tid + j * 128;
        if (j < 2) {            // K part: chunks [0,256), ldsK contiguous
          int kr = cidx >> 3, part = cidx & 7;
          copy16(K + (krow + s0 + kr) * 512 + h * DK + part * 8,
                 &ldsK[buf][cidx * 8]);
        } else {                // V part: chunks [0,256) of ldsV
          int c2 = cidx - 256;
          int vr = c2 >> 2, part = c2 & 3;
          copy16(Vt + (vtbase + vr) * 512 + s0 + part * 8,
                 &ldsV[buf][c2 * 8]);
        }
      }
    };

    __syncthreads();              // protect buffers from previous c's readers
    issue(0, 0);

    for (int it = 0; it < TT / 32; ++it) {
      const int buf = it & 1;
      wait_async();
      __syncthreads();            // tile `it` visible to all waves
      if (it + 1 < TT / 32) issue((it + 1) * 32, buf ^ 1);

      const bf16_t* bk = &ldsK[buf][0];
      const bf16_t* bv = &ldsV[buf][0];

      // S = Q(16x64) * K^T(64x32)
      v8f z = {};
      v8f S0 = wmma_bf16(qf0, bfrag(bk + (size_t)l16 * 64 + 16 * half), z);
      S0 = wmma_bf16(qf1, bfrag(bk + (size_t)l16 * 64 + 16 * half + 32), S0);
      v8f S1 = wmma_bf16(qf0, bfrag(bk + (size_t)(16 + l16) * 64 + 16 * half), z);
      S1 = wmma_bf16(qf1, bfrag(bk + (size_t)(16 + l16) * 64 + 16 * half + 32), S1);

      float corr[8];
#pragma unroll
      for (int r = 0; r < 8; ++r) {
        float a0 = S0[r] * 0.125f, a1 = S1[r] * 0.125f;
        float t = fmaxf(a0, a1);
        t = fmaxf(t, __shfl_xor(t, 1, 32));
        t = fmaxf(t, __shfl_xor(t, 2, 32));
        t = fmaxf(t, __shfl_xor(t, 4, 32));
        t = fmaxf(t, __shfl_xor(t, 8, 32));
        const float mn = fmaxf(mrow[r], t);
        corr[r] = __expf(mrow[r] - mn);
        const float p0 = __expf(a0 - mn);
        const float p1 = __expf(a1 - mn);
        S0[r] = p0; S1[r] = p1;
        float s = p0 + p1;
        s += __shfl_xor(s, 1, 32);
        s += __shfl_xor(s, 2, 32);
        s += __shfl_xor(s, 4, 32);
        s += __shfl_xor(s, 8, 32);
        lrow[r] = lrow[r] * corr[r] + s;
        mrow[r] = mn;
      }

#pragma unroll
      for (int r = 0; r < 8; ++r) {
        const int m = r + 8 * half;
        pbase[m * PSTRIDE + l16]      = (bf16_t)S0[r];
        pbase[m * PSTRIDE + 16 + l16] = (bf16_t)S1[r];
      }
      __syncthreads();            // uniform; also orders per-wave pshm write->read
      const v16bf pf = afrag(pbase + l16 * PSTRIDE + 16 * half);

#pragma unroll
      for (int r = 0; r < 8; ++r) {
        O0[r] *= corr[r]; O1[r] *= corr[r]; O2[r] *= corr[r]; O3[r] *= corr[r];
      }
      O0 = wmma_bf16(pf, bfrag(bv + (size_t)l16 * 32 + 16 * half), O0);
      O1 = wmma_bf16(pf, bfrag(bv + (size_t)(16 + l16) * 32 + 16 * half), O1);
      O2 = wmma_bf16(pf, bfrag(bv + (size_t)(32 + l16) * 32 + 16 * half), O2);
      O3 = wmma_bf16(pf, bfrag(bv + (size_t)(48 + l16) * 32 + 16 * half), O3);
    }
#pragma unroll
    for (int r = 0; r < 8; ++r) {
      const float inv = 1.0f / lrow[r];
      acc0[r] += O0[r] * inv; acc1[r] += O1[r] * inv;
      acc2[r] += O2[r] * inv; acc3[r] += O3[r] * inv;
    }
  }

  // 1/N average over key variables, emit bf16 ctx [N*B*T, 512]
#pragma unroll
  for (int r = 0; r < 8; ++r) {
    const int m = r + 8 * half;
    const size_t off = (qrow + m) * 512 + h * DK;
    ctxh[off +      l16] = (bf16_t)(acc0[r] * 0.25f);
    ctxh[off + 16 + l16] = (bf16_t)(acc1[r] * 0.25f);
    ctxh[off + 32 + l16] = (bf16_t)(acc2[r] * 0.25f);
    ctxh[off + 48 + l16] = (bf16_t)(acc3[r] * 0.25f);
  }
}

// ---------- residual + LayerNorm ----------
__global__ void ln_kernel(const float* __restrict__ x, const float* __restrict__ proj,
                          const float* __restrict__ gamma, const float* __restrict__ beta,
                          float* __restrict__ out) {
  const int row = blockIdx.x;
  const int tid = threadIdx.x;
  const float* xr = x + (size_t)row * 512;
  const float* pr = proj + (size_t)row * 512;
  const float v0 = xr[tid] + pr[tid];
  const float v1 = xr[tid + 256] + pr[tid + 256];
  float s = v0 + v1, ss = v0 * v0 + v1 * v1;
#pragma unroll
  for (int m = 1; m < 32; m <<= 1) {
    s  += __shfl_xor(s, m, 32);
    ss += __shfl_xor(ss, m, 32);
  }
  __shared__ float sb[8], qb[8], tot[2];
  const int w = tid >> 5;
  if ((tid & 31) == 0) { sb[w] = s; qb[w] = ss; }
  __syncthreads();
  if (tid == 0) {
    float a = 0.f, cc = 0.f;
    for (int i = 0; i < 8; ++i) { a += sb[i]; cc += qb[i]; }
    tot[0] = a; tot[1] = cc;
  }
  __syncthreads();
  const float mean = tot[0] * (1.0f / 512.0f);
  const float var  = tot[1] * (1.0f / 512.0f) - mean * mean;
  const float rstd = rsqrtf(var + 1e-5f);
  out[(size_t)row * 512 + tid]       = (v0 - mean) * rstd * gamma[tid] + beta[tid];
  out[(size_t)row * 512 + tid + 256] = (v1 - mean) * rstd * gamma[tid + 256] + beta[tid + 256];
}

extern "C" void kernel_launch(void* const* d_in, const int* in_sizes, int n_in,
                              void* d_out, int out_size, void* d_ws, size_t ws_size,
                              hipStream_t stream) {
  const float* x     = (const float*)d_in[0];
  const float* Wq    = (const float*)d_in[1];
  const float* bq    = (const float*)d_in[2];
  const float* Wk    = (const float*)d_in[3];
  const float* bk    = (const float*)d_in[4];
  const float* Wv    = (const float*)d_in[5];
  const float* bv    = (const float*)d_in[6];
  const float* Wo    = (const float*)d_in[7];
  const float* bo    = (const float*)d_in[8];
  const float* gamma = (const float*)d_in[9];
  const float* beta  = (const float*)d_in[10];
  float* out = (float*)d_out;

  const size_t NBT = (size_t)NV * BB * TT;          // 8192
  const size_t ACT = NBT * 512;                     // 4,194,304 elements
  const size_t WEL = 512 * 512;

  char* ws = (char*)d_ws;
  size_t off = 0;
  bf16_t* xh  = (bf16_t*)(ws + off); off += ACT * 2;
  bf16_t* WqT = (bf16_t*)(ws + off); off += WEL * 2;
  bf16_t* WkT = (bf16_t*)(ws + off); off += WEL * 2;
  bf16_t* WvT = (bf16_t*)(ws + off); off += WEL * 2;
  bf16_t* WoT = (bf16_t*)(ws + off); off += WEL * 2;
  bf16_t* Qb  = (bf16_t*)(ws + off); off += ACT * 2;
  bf16_t* Kb  = (bf16_t*)(ws + off); off += ACT * 2;
  bf16_t* Vt  = (bf16_t*)(ws + off); off += ACT * 2;
  bf16_t* ctxh= (bf16_t*)(ws + off); off += ACT * 2;
  float*  proj= (float*)(ws + off);  off += ACT * 4;

  cvt_bf16<<<(int)((ACT + 255) / 256), 256, 0, stream>>>(x, xh, (int)ACT);
  cvt_bf16_T<<<1024, 256, 0, stream>>>(Wq, WqT);
  cvt_bf16_T<<<1024, 256, 0, stream>>>(Wk, WkT);
  cvt_bf16_T<<<1024, 256, 0, stream>>>(Wv, WvT);
  cvt_bf16_T<<<1024, 256, 0, stream>>>(Wo, WoT);

  const int gemm_grid = (int)(NBT / 32) * 8;        // 2048 waves
  gemm_bf16<<<gemm_grid, 32, 0, stream>>>(xh, WqT, bq, Qb, 0);
  gemm_bf16<<<gemm_grid, 32, 0, stream>>>(xh, WkT, bk, Kb, 0);
  gemm_bf16<<<gemm_grid, 32, 0, stream>>>(xh, WvT, bv, Vt, 1);

  attn_kernel<<<NV * BB * HH * (TT / 64), 128, 0, stream>>>(Qb, Kb, Vt, ctxh);

  gemm_bf16<<<gemm_grid, 32, 0, stream>>>(ctxh, WoT, bo, proj, 2);

  ln_kernel<<<(int)NBT, 256, 0, stream>>>(x, proj, gamma, beta, out);
}
